// FedSimModel_78314433675390
// MI455X (gfx1250) — matrix operand order repaired and checked by reading, not compiled, not run on hardware
//
#include <hip/hip_runtime.h>
#include <math.h>

#define B_ROWS 16384
#define D_DIM  512
#define K_TXT  5

typedef __attribute__((ext_vector_type(16))) __bf16 v16bf;
typedef __attribute__((ext_vector_type(8)))  float  v8f;

union FragBF { unsigned int u[8]; v16bf v; };

// Round-to-nearest f32->bf16 pair pack: add 0x8000 then byte-permute the two
// high halves together. 3 VALU ops total (2x v_add_u32 + 1x v_perm_b32).
__device__ __forceinline__ unsigned int bf16_pk(float lo, float hi) {
    unsigned int ul = __float_as_uint(lo) + 0x8000u;
    unsigned int uh = __float_as_uint(hi) + 0x8000u;
    return __builtin_amdgcn_perm(uh, ul, 0x07060302u);
}
__device__ __forceinline__ float wave_sum(float v) {
#pragma unroll
    for (int m = 16; m > 0; m >>= 1) v += __shfl_xor(v, m, 32);
    return v;
}
// Branch-free tanh-approx GELU: x * sigmoid(2 * 0.79788456*(x + 0.044715 x^3)).
__device__ __forceinline__ float gelu_tanh(float x) {
    float u = 0.7978845608028654f * (x + 0.044715f * x * x * x);
    return x / (1.f + __expf(-2.f * u));
}
__device__ __forceinline__ float sigmoid_fast(float x) {
    return 1.f / (1.f + __expf(-x));
}

// ---------------------------------------------------------------------------
// Pack fp32 weight [512(K) x 512(N)] row-major into bf16 B-operand fragments,
// folding the BatchNorm per-K scale: W'[k,n] = scale[k] * W[k,n].
// Fragment (ntile, kt) covers K=kt*32..+31, N=ntile*16..+15.
// Per-lane layout: lane L holds col n = ntile*16 + (L&15),
//   K-half base = kt*32 + (L>>4)*16, dword v holds K = base+2v (lo), +1 (hi).
// Stored so each lane's 8 dwords are contiguous: idx = ((frag*32 + L)*8 + v).
// ---------------------------------------------------------------------------
__global__ __launch_bounds__(256) void pack_w_scaled(const float* __restrict__ W,
                                                     const float* __restrict__ stat,
                                                     unsigned int* __restrict__ out) {
    int i = blockIdx.x * 256 + threadIdx.x;       // 0 .. 131071
    int v    = i & 7;
    int lane = (i >> 3) & 31;
    int frag = i >> 8;                            // ntile*16 + kt
    int kt = frag & 15, ntile = frag >> 4;
    int n = ntile * 16 + (lane & 15);
    int k = kt * 32 + (lane >> 4) * 16 + v * 2;
    out[i] = bf16_pk(W[k * D_DIM + n] * stat[k],
                     W[(k + 1) * D_DIM + n] * stat[k + 1]);
}

// bias'[n] = bias[n] + sum_k shift[k] * W[k,n]   (fixed-order, deterministic)
__global__ __launch_bounds__(256) void fold_bias(const float* __restrict__ W,
                                                 const float* __restrict__ stat,
                                                 const float* __restrict__ bias_in,
                                                 float* __restrict__ bias_out) {
    const int n = blockIdx.x * 256 + threadIdx.x;   // 512 outputs
    float acc = bias_in[n];
    for (int k = 0; k < D_DIM; ++k)
        acc += stat[D_DIM + k] * W[k * D_DIM + n];
    bias_out[n] = acc;
}

// ---------------------------------------------------------------------------
// Front end: cosine sim -> weight-gate MLP -> rank sort -> conv merge.
// One wave per row.
// ---------------------------------------------------------------------------
__global__ __launch_bounds__(256) void merge_kernel(
    const float* __restrict__ img, const float* __restrict__ txt,
    const float* __restrict__ w1, const float* __restrict__ b1,
    const float* __restrict__ w2, const float* __restrict__ b2,
    const float* __restrict__ cw, const float* __restrict__ cb,
    float* __restrict__ X) {
    const int lane = threadIdx.x & 31, wid = threadIdx.x >> 5;
    const int b = blockIdx.x * 8 + wid;
    const float* ir = img + (size_t)b * D_DIM;
    float4 iq[4], tq[K_TXT][4];
    float img2 = 0.f, dot[K_TXT], t2[K_TXT];
#pragma unroll
    for (int p = 0; p < 4; ++p) {
        iq[p] = *(const float4*)(ir + lane * 4 + p * 128);
        img2 += iq[p].x * iq[p].x + iq[p].y * iq[p].y + iq[p].z * iq[p].z + iq[p].w * iq[p].w;
    }
#pragma unroll
    for (int k = 0; k < K_TXT; ++k) {
        const float* tr = txt + ((size_t)b * K_TXT + k) * D_DIM;
        dot[k] = 0.f; t2[k] = 0.f;
#pragma unroll
        for (int p = 0; p < 4; ++p) {
            float4 t = *(const float4*)(tr + lane * 4 + p * 128);
            tq[k][p] = t;
            dot[k] += iq[p].x * t.x + iq[p].y * t.y + iq[p].z * t.z + iq[p].w * t.w;
            t2[k]  += t.x * t.x + t.y * t.y + t.z * t.z + t.w * t.w;
        }
    }
    img2 = wave_sum(img2);
    float ni = 1.f / fmaxf(sqrtf(img2), 1e-12f);
    float w[K_TXT];
#pragma unroll
    for (int k = 0; k < K_TXT; ++k) {
        dot[k] = wave_sum(dot[k]);
        t2[k]  = wave_sum(t2[k]);
        float sim = dot[k] * ni / fmaxf(sqrtf(t2[k]), 1e-12f);
        float acc = b2[0];
#pragma unroll
        for (int j = 0; j < 16; ++j)
            acc += fmaxf(sim * w1[j] + b1[j], 0.f) * w2[j];
        w[k] = sigmoid_fast(acc);
    }
    float coef[K_TXT];
#pragma unroll
    for (int k = 0; k < K_TXT; ++k) {
        int r = 0;
#pragma unroll
        for (int j = 0; j < K_TXT; ++j)
            r += (w[j] > w[k]) || (w[j] == w[k] && j < k);  // stable desc rank
        coef[k] = w[k] * cw[r];
    }
    const float cbv = cb[0];
    float* xo = X + (size_t)b * D_DIM;
#pragma unroll
    for (int p = 0; p < 4; ++p) {
        float4 o; o.x = cbv; o.y = cbv; o.z = cbv; o.w = cbv;
#pragma unroll
        for (int k = 0; k < K_TXT; ++k) {
            o.x += coef[k] * tq[k][p].x; o.y += coef[k] * tq[k][p].y;
            o.z += coef[k] * tq[k][p].z; o.w += coef[k] * tq[k][p].w;
        }
        *(float4*)(xo + lane * 4 + p * 128) = o;
    }
}

// ---------------------------------------------------------------------------
// Deterministic two-stage BatchNorm stats (no float atomics).
// ---------------------------------------------------------------------------
__global__ __launch_bounds__(256) void stats_partial(const float* __restrict__ X,
                                                     float* __restrict__ part) {
    const int t = threadIdx.x, blk = blockIdx.x;     // 128 blocks x 128 rows
    const float* base = X + (size_t)blk * 128 * D_DIM;
    float s0 = 0.f, q0 = 0.f, s1 = 0.f, q1 = 0.f;
    const int f0 = t, f1 = t + 256;
    for (int r = 0; r < 128; ++r) {
        float a = base[r * D_DIM + f0];
        float b = base[r * D_DIM + f1];
        s0 += a; q0 += a * a; s1 += b; q1 += b * b;
    }
    part[(blk * 2 + 0) * D_DIM + f0] = s0;
    part[(blk * 2 + 1) * D_DIM + f0] = q0;
    part[(blk * 2 + 0) * D_DIM + f1] = s1;
    part[(blk * 2 + 1) * D_DIM + f1] = q1;
}

__global__ __launch_bounds__(256) void stats_reduce(const float* __restrict__ part,
                                                    const float* __restrict__ g,
                                                    const float* __restrict__ bb,
                                                    float* __restrict__ stat) {
    const int f = blockIdx.x * 256 + threadIdx.x;    // 512 features
    float m = 0.f, q = 0.f;
    for (int blk = 0; blk < 128; ++blk) {
        m += part[(blk * 2 + 0) * D_DIM + f];
        q += part[(blk * 2 + 1) * D_DIM + f];
    }
    float mean = m * (1.f / (float)B_ROWS);
    float var  = q * (1.f / (float)B_ROWS) - mean * mean;   // biased, matches jnp.var
    float sc   = rsqrtf(var + 1e-5f) * g[f];
    stat[f]          = sc;                 // scale  (folded into W)
    stat[D_DIM + f]  = bb[f] - mean * sc;  // shift  (folded into bias)
}

// ---------------------------------------------------------------------------
// GEMM: Y[B,512] = act( A @ W' + bias' ), bf16 WMMA, f32 accumulate.
// BN affine pre-folded into W'/bias'. Wave: 16 rows x 128 cols (8 acc tiles).
// Block: 8 waves = 128 rows, one N-group. Grid: (B/128)*(512/128) = 512 blocks.
// All 16 B-fragment loads of a K-step are issued before the WMMA chain so the
// scheduler can stagger s_wait_loadcnt instead of draining per-WMMA; next
// K-step's raw A float4s are prefetched under the WMMA chain.
// ---------------------------------------------------------------------------
template <int ACT>
__global__ __launch_bounds__(256) void gemm512(const float* __restrict__ A,
                                               const unsigned int* __restrict__ Wp,
                                               const float* __restrict__ bias,
                                               float* __restrict__ Y) {
    const int lane = threadIdx.x & 31;
    const int wid  = threadIdx.x >> 5;
    const int mb   = blockIdx.x >> 2;
    const int ng   = blockIdx.x & 3;
    const int mbase = (mb * 8 + wid) * 16;
    const int row   = mbase + (lane & 15);
    const int khalf = (lane >> 4) * 16;         // A-operand K sub-base for this lane
    const float* arow = A + (size_t)row * D_DIM;
    // Base of this wave's B fragments: frag index = (ng*8 + nt)*16 + kt.
    const unsigned int* wbase = Wp + (size_t)(ng * 8 * 16 * 32 * 8) + (size_t)lane * 8;

    v8f c[8];
#pragma unroll
    for (int i = 0; i < 8; ++i)
        c[i] = (v8f){0.f, 0.f, 0.f, 0.f, 0.f, 0.f, 0.f, 0.f};

    // Prime the A pipeline with kt=0's raw row data.
    float4 qa0 = *(const float4*)(arow + khalf);
    float4 qa1 = *(const float4*)(arow + khalf + 4);
    float4 qa2 = *(const float4*)(arow + khalf + 8);
    float4 qa3 = *(const float4*)(arow + khalf + 12);

    for (int kt = 0; kt < 16; ++kt) {
        // Issue all 16 B-fragment b128 loads for this K-step up front.
        FragBF bf[8];
#pragma unroll
        for (int nt = 0; nt < 8; ++nt) {
            const unsigned int* fp = wbase + (size_t)((nt * 16 + kt) * 32) * 8;
            *(uint4*)&bf[nt].u[0] = *(const uint4*)(fp);
            *(uint4*)&bf[nt].u[4] = *(const uint4*)(fp + 4);
        }
        // Pack current A fragment (VALU overlaps the in-flight loads).
        FragBF a;
        a.u[0] = bf16_pk(qa0.x, qa0.y); a.u[1] = bf16_pk(qa0.z, qa0.w);
        a.u[2] = bf16_pk(qa1.x, qa1.y); a.u[3] = bf16_pk(qa1.z, qa1.w);
        a.u[4] = bf16_pk(qa2.x, qa2.y); a.u[5] = bf16_pk(qa2.z, qa2.w);
        a.u[6] = bf16_pk(qa3.x, qa3.y); a.u[7] = bf16_pk(qa3.z, qa3.w);
        // Prefetch next K-step's raw A row chunk under the WMMA chain.
        if (kt < 15) {
            const int kb = (kt + 1) * 32 + khalf;
            qa0 = *(const float4*)(arow + kb);
            qa1 = *(const float4*)(arow + kb + 4);
            qa2 = *(const float4*)(arow + kb + 8);
            qa3 = *(const float4*)(arow + kb + 12);
        }
#pragma unroll
        for (int nt = 0; nt < 8; ++nt) {
            c[nt] = __builtin_amdgcn_wmma_f32_16x16x32_bf16(
                false, a.v, false, bf[nt].v, (short)0, c[nt], false, false);
        }
    }
    // Epilogue: bias + activation, fp32 store.
    const int colbase = ng * 128 + (lane & 15);
    const int rbase   = (lane >> 4) * 8;
#pragma unroll
    for (int nt = 0; nt < 8; ++nt) {
        const int col = colbase + nt * 16;
        const float bn = bias[col];
#pragma unroll
        for (int r = 0; r < 8; ++r) {
            float v = c[nt][r] + bn;
            if (ACT) v = gelu_tanh(v);
            Y[(size_t)(mbase + rbase + r) * D_DIM + col] = v;
        }
    }
}

// ---------------------------------------------------------------------------
// Row-wise L2 normalize (no eps, matches reference final step). Wave per row.
// ---------------------------------------------------------------------------
__global__ __launch_bounds__(256) void l2norm_kernel(const float* __restrict__ Yin,
                                                     float* __restrict__ Out) {
    const int lane = threadIdx.x & 31, wid = threadIdx.x >> 5;
    const int b = blockIdx.x * 8 + wid;
    const float* yr = Yin + (size_t)b * D_DIM;
    float4 q[4]; float s = 0.f;
#pragma unroll
    for (int p = 0; p < 4; ++p) {
        q[p] = *(const float4*)(yr + lane * 4 + p * 128);
        s += q[p].x * q[p].x + q[p].y * q[p].y + q[p].z * q[p].z + q[p].w * q[p].w;
    }
    s = wave_sum(s);
    const float inv = 1.f / sqrtf(s);
    float* o = Out + (size_t)b * D_DIM;
#pragma unroll
    for (int p = 0; p < 4; ++p) {
        float4 r; r.x = q[p].x * inv; r.y = q[p].y * inv;
        r.z = q[p].z * inv; r.w = q[p].w * inv;
        *(float4*)(o + lane * 4 + p * 128) = r;
    }
}

extern "C" void kernel_launch(void* const* d_in, const int* in_sizes, int n_in,
                              void* d_out, int out_size, void* d_ws, size_t ws_size,
                              hipStream_t stream) {
    (void)in_sizes; (void)n_in; (void)out_size; (void)ws_size;
    const float* img    = (const float*)d_in[0];
    const float* txt    = (const float*)d_in[1];
    const float* wg_w1  = (const float*)d_in[2];
    const float* wg_b1  = (const float*)d_in[3];
    const float* wg_w2  = (const float*)d_in[4];
    const float* wg_b2  = (const float*)d_in[5];
    const float* conv_w = (const float*)d_in[6];
    const float* conv_b = (const float*)d_in[7];
    const float* mg_bn_g = (const float*)d_in[8];
    const float* mg_bn_b = (const float*)d_in[9];
    const float* mg_w    = (const float*)d_in[10];
    const float* mg_b    = (const float*)d_in[11];
    const float* ph_bn1_g = (const float*)d_in[12];
    const float* ph_bn1_b = (const float*)d_in[13];
    const float* ph_w1    = (const float*)d_in[14];
    const float* ph_b1    = (const float*)d_in[15];
    const float* ph_bn2_g = (const float*)d_in[16];
    const float* ph_bn2_b = (const float*)d_in[17];
    const float* ph_w2    = (const float*)d_in[18];
    const float* ph_b2    = (const float*)d_in[19];

    const size_t NBD = (size_t)B_ROWS * D_DIM;
    const size_t PWD = (size_t)D_DIM * D_DIM / 2;   // dwords per packed weight
    float* ws = (float*)d_ws;
    float* x_buf  = ws;                    // 32 MB, later reused as y_img
    float* fusedb = ws + NBD;              // 32 MB, later reused as y_fused
    unsigned int* pwbase = (unsigned int*)(ws + 2 * NBD);
    unsigned int* pw0 = pwbase + 0 * PWD;  // 0.5 MB each, 5 folded weight packs
    unsigned int* pw1 = pwbase + 1 * PWD;
    unsigned int* pw2 = pwbase + 2 * PWD;
    unsigned int* pw3 = pwbase + 3 * PWD;
    unsigned int* pw4 = pwbase + 4 * PWD;
    float* biasf = (float*)(pwbase + 5 * PWD);      // 5 folded bias vectors
    float* bias0 = biasf + 0 * D_DIM;
    float* bias1 = biasf + 1 * D_DIM;
    float* bias2 = biasf + 2 * D_DIM;
    float* bias3 = biasf + 3 * D_DIM;
    float* bias4 = biasf + 4 * D_DIM;
    float* stats = biasf + 5 * D_DIM;
    float* stat_x     = stats;
    float* stat_img   = stats + 1024;
    float* stat_fused = stats + 2048;
    float* stat_z1i   = stats + 3072;
    float* stat_z1f   = stats + 4096;
    float* part       = stats + 5120;      // 128*2*512 floats

    float* out_img   = (float*)d_out;      // also z1_img scratch
    float* out_fused = out_img + NBD;      // also z1_fused scratch

    // Front end: sim -> gate -> sort -> conv merge.
    merge_kernel<<<B_ROWS / 8, 256, 0, stream>>>(img, txt, wg_w1, wg_b1, wg_w2,
                                                 wg_b2, conv_w, conv_b, x_buf);

    // BN stats for merge-gate input and for the image branch.
    stats_partial<<<128, 256, 0, stream>>>(x_buf, part);
    stats_reduce<<<2, 256, 0, stream>>>(part, mg_bn_g, mg_bn_b, stat_x);
    stats_partial<<<128, 256, 0, stream>>>(img, part);
    stats_reduce<<<2, 256, 0, stream>>>(part, ph_bn1_g, ph_bn1_b, stat_img);

    // fused = GELU(BN(x) @ mg_w + mg_b), BN folded into pw0/bias0.
    pack_w_scaled<<<512, 256, 0, stream>>>(mg_w, stat_x, pw0);
    fold_bias<<<2, 256, 0, stream>>>(mg_w, stat_x, mg_b, bias0);
    gemm512<1><<<512, 256, 0, stream>>>(x_buf, pw0, bias0, fusedb);
    stats_partial<<<128, 256, 0, stream>>>(fusedb, part);
    stats_reduce<<<2, 256, 0, stream>>>(part, ph_bn1_g, ph_bn1_b, stat_fused);

    // z1 = GELU(BN1(z) @ ph_w1 + b1) for both branches (z1 parked in d_out).
    pack_w_scaled<<<512, 256, 0, stream>>>(ph_w1, stat_img, pw1);
    fold_bias<<<2, 256, 0, stream>>>(ph_w1, stat_img, ph_b1, bias1);
    gemm512<1><<<512, 256, 0, stream>>>(img, pw1, bias1, out_img);

    pack_w_scaled<<<512, 256, 0, stream>>>(ph_w1, stat_fused, pw2);
    fold_bias<<<2, 256, 0, stream>>>(ph_w1, stat_fused, ph_b1, bias2);
    gemm512<1><<<512, 256, 0, stream>>>(fusedb, pw2, bias2, out_fused);

    stats_partial<<<128, 256, 0, stream>>>(out_img, part);
    stats_reduce<<<2, 256, 0, stream>>>(part, ph_bn2_g, ph_bn2_b, stat_z1i);
    stats_partial<<<128, 256, 0, stream>>>(out_fused, part);
    stats_reduce<<<2, 256, 0, stream>>>(part, ph_bn2_g, ph_bn2_b, stat_z1f);

    // y = BN2(z1) @ ph_w2 + b2, then row L2-normalize into d_out.
    pack_w_scaled<<<512, 256, 0, stream>>>(ph_w2, stat_z1i, pw3);
    fold_bias<<<2, 256, 0, stream>>>(ph_w2, stat_z1i, ph_b2, bias3);
    gemm512<0><<<512, 256, 0, stream>>>(out_img, pw3, bias3, x_buf);

    pack_w_scaled<<<512, 256, 0, stream>>>(ph_w2, stat_z1f, pw4);
    fold_bias<<<2, 256, 0, stream>>>(ph_w2, stat_z1f, ph_b2, bias4);
    gemm512<0><<<512, 256, 0, stream>>>(out_fused, pw4, bias4, fusedb);

    l2norm_kernel<<<B_ROWS / 8, 256, 0, stream>>>(x_buf,  out_img);
    l2norm_kernel<<<B_ROWS / 8, 256, 0, stream>>>(fusedb, out_fused);
}
